// BigBirdSeq2Seq_69020124446913
// MI455X (gfx1250) — matrix operand order repaired and checked by reading, not compiled, not run on hardware
//
#include <hip/hip_runtime.h>
#include <hip/hip_bf16.h>

// ---------------------------------------------------------------------------
// BigBird encoder forward for MI455X (gfx1250, wave32, WMMA).
// All GEMMs + attention use v_wmma_f32_16x16x32_bf16 (bf16 in, f32 accum).
// B operands are staged into LDS with K-pairs pre-packed into dwords so every
// fragment load is a plain ds_load_b32/b128 (no inner-loop repack VALU).
// Staging issues all global loads into independent temps before any LDS store
// so the compiler drains LOADcnt incrementally instead of serializing.
// ---------------------------------------------------------------------------

typedef unsigned short u16;
typedef __bf16 v16bf __attribute__((ext_vector_type(16)));
typedef float  v8f   __attribute__((ext_vector_type(8)));
typedef unsigned int v8u __attribute__((ext_vector_type(8)));

constexpr int B_ = 4, L_ = 2048, E_ = 512, H_ = 8, DH_ = 64, M_ = 2048;
constexpr int NL_ = 4, NB_ = 32, NR_ = 3;   // NB_ = L_/64 key blocks per row

// ----------------------------- helpers -------------------------------------

__device__ __forceinline__ u16 f2bf(float f) {
  unsigned u = __builtin_bit_cast(unsigned, f);
  u += 0x7fffu + ((u >> 16) & 1u);          // round-to-nearest-even
  return (u16)(u >> 16);
}

__device__ __forceinline__ unsigned pack2bf(float a, float b) {
  return (unsigned)f2bf(a) | ((unsigned)f2bf(b) << 16);
}

__device__ __forceinline__ v8f v8f_zero() {
  v8f z = {0.f, 0.f, 0.f, 0.f, 0.f, 0.f, 0.f, 0.f};
  return z;
}

__device__ __forceinline__ v8f wmma_bf16(v16bf a, v16bf b, v8f c) {
  // (neg_a, A, neg_b, B, c_mod, C, reuse_a, reuse_b)
  return __builtin_amdgcn_wmma_f32_16x16x32_bf16(false, a, false, b,
                                                 (short)0, c, false, false);
}

// A fragment (16x32 MxK) from LDS u16 row-major tile with element stride.
// Layout used here (contiguous k): lane m = l&15, half = l>>4; VGPR r holds K pair
//   k = r*2 + half*16   (8 contiguous pairs)
__device__ __forceinline__ v16bf load_a_frag(const u16* base, int stride) {
  unsigned lane = threadIdx.x & 31u;
  unsigned m = lane & 15u, hf = lane >> 4;
  v8u u;
#pragma unroll
  for (int r = 0; r < 8; ++r) {
    unsigned k = (unsigned)r * 2u + hf * 16u;
    u[r] = *(const unsigned*)(base + (size_t)m * stride + k);
  }
  return __builtin_bit_cast(v16bf, u);
}

// B fragment (32x16 KxN) from pair-packed LDS: base[kp * stride + n] holds
// (B[2kp][n], B[2kp+1][n]). This code uses VGPR r = pair k = r*2 + (l>>4)*16,
// i.e. kp = r + (l>>4)*8 -> single b32 read per VGPR.
__device__ __forceinline__ v16bf load_bp_frag(const unsigned* base, int stride) {
  unsigned lane = threadIdx.x & 31u;
  unsigned n = lane & 15u, hf = lane >> 4;
  v8u u;
#pragma unroll
  for (int r = 0; r < 8; ++r) {
    unsigned kp = (unsigned)r + hf * 8u;
    u[r] = base[(size_t)kp * stride + n];
  }
  return __builtin_bit_cast(v16bf, u);
}

// B fragment from transposed u16 storage [n][k] (k contiguous -> b32 reads).
__device__ __forceinline__ v16bf load_bt_frag(const u16* base, int stride) {
  unsigned lane = threadIdx.x & 31u;
  unsigned n = lane & 15u, hf = lane >> 4;
  v8u u;
#pragma unroll
  for (int r = 0; r < 8; ++r) {
    unsigned k = (unsigned)r * 2u + hf * 16u;
    u[r] = *(const unsigned*)(base + (size_t)n * stride + k);
  }
  return __builtin_bit_cast(v16bf, u);
}

__device__ __forceinline__ float gelu_f(float x) {
  float x3 = x * x * x;
  return 0.5f * x * (1.f + tanhf(0.7978845608028654f * (x + 0.044715f * x3)));
}

// ----------------------------- small kernels --------------------------------

// Per-layer BigBird block mask as one 32-bit word per query block row.
__global__ void build_masks_kernel(unsigned* __restrict__ masks) {
  int idx = threadIdx.x;
  if (idx >= NL_ * NB_) return;
  int lyr = idx >> 5, i = idx & 31;
  unsigned m = 0;
  if (i > 0) m |= 1u << (i - 1);
  m |= 1u << i;
  if (i < NB_ - 1) m |= 1u << (i + 1);
  m |= 1u;                 // global first block col
  m |= 1u << (NB_ - 1);    // global last block col
  if (i == 0 || i == NB_ - 1) {
    m = 0xffffffffu;       // global rows attend everywhere
  } else {
    unsigned avoid = 1u | (1u << (NB_ - 1)) | (1u << (i - 1)) | (1u << i) | (1u << (i + 1));
    unsigned state = 0x9E3779B9u * (unsigned)(lyr * 131 + i + 1);
    unsigned chosen = 0;
    int picked = 0;
    for (int attempt = 0; attempt < 1024 && picked < NR_; ++attempt) {
      state = state * 1664525u + 1013904223u;
      unsigned j = (state >> 11) & 31u;
      if ((avoid >> j) & 1u) continue;
      if ((chosen >> j) & 1u) continue;
      chosen |= 1u << j;
      m |= 1u << j;
      ++picked;
    }
  }
  masks[idx] = m;
}

// f32 -> bf16 matrix copy with output stride / column offset (for packing
// wq|wk|wv into one [E, 3E] operand).
__launch_bounds__(256)
__global__ void f32_to_bf16_mat(const float* __restrict__ src, u16* __restrict__ dst,
                                int rows, int cols, int dst_stride, int col_off) {
  int i = blockIdx.x * 256 + threadIdx.x;
  if (i >= rows * cols) return;
  int r = i / cols, c = i - r * cols;
  dst[(size_t)r * dst_stride + col_off + c] = f2bf(src[i]);
}

// x[b,l,:] = embed[tokens[b,l]] + sinusoidal PE(l)
__launch_bounds__(128)
__global__ void embed_pe_kernel(const int* __restrict__ tokens,
                                const float* __restrict__ emb,
                                float* __restrict__ x) {
  const int bl = blockIdx.x;
  const int pos = bl & (L_ - 1);
  const int tok = tokens[bl];
  const int t = threadIdx.x;
  const float c0 = -logf(10000.f) / 256.f;
#pragma unroll
  for (int i = 0; i < 4; ++i) {
    int ch = t + i * 128;
    float pe;
    if (ch < 256) pe = __sinf((float)pos * __expf((float)ch * c0));
    else          pe = __cosf((float)pos * __expf((float)(ch - 256) * c0));
    x[(size_t)bl * E_ + ch] = emb[(size_t)tok * E_ + ch] + pe;
  }
}

__device__ __forceinline__ float block_sum_128(float v, float* sh) {
#pragma unroll
  for (int off = 16; off; off >>= 1) v += __shfl_xor(v, off, 32);
  __syncthreads();
  if ((threadIdx.x & 31u) == 0) sh[threadIdx.x >> 5] = v;
  __syncthreads();
  return sh[0] + sh[1] + sh[2] + sh[3];
}

// LayerNorm over E=512; optional f32 and/or bf16 output.
__launch_bounds__(128)
__global__ void layernorm_kernel(const float* __restrict__ x, const float* __restrict__ g,
                                 const float* __restrict__ bb, float* __restrict__ outF,
                                 u16* __restrict__ outB) {
  __shared__ float sh[4];
  const int row = blockIdx.x;
  const float* xr = x + (size_t)row * E_;
  const int t = threadIdx.x;
  float v[4];
  float s = 0.f;
#pragma unroll
  for (int i = 0; i < 4; ++i) { v[i] = xr[t + i * 128]; s += v[i]; }
  s = block_sum_128(s, sh);
  float mu = s * (1.f / (float)E_);
  float q = 0.f;
#pragma unroll
  for (int i = 0; i < 4; ++i) { float d = v[i] - mu; q += d * d; }
  q = block_sum_128(q, sh);
  float inv = rsqrtf(q * (1.f / (float)E_) + 1e-6f);
#pragma unroll
  for (int i = 0; i < 4; ++i) {
    int c = t + i * 128;
    float o = (v[i] - mu) * inv * g[c] + bb[c];
    if (outF) outF[(size_t)row * E_ + c] = o;
    if (outB) outB[(size_t)row * E_ + c] = f2bf(o);
  }
}

// ----------------------------- WMMA GEMM ------------------------------------
// C[256x64 tile] = A[M x K](bf16) * B[K x N](bf16), f32 accumulate.
// 8 waves; wave w owns rows w*32..w*32+31 (2 M-subtiles), all 4 N-tiles.
// 8 WMMA / wave / K-step; A frags reused x4, B frags reused x2.
// mode 0: qkv  -> outF = acc * (col<512 ? 1/sqrt(DH) : 1)
// mode 1: oproj-> outF = residual + acc
// mode 2: mlp1 -> outB = bf16(gelu(acc + bias[col]))
// mode 3: mlp2 -> outF = residual + acc + bias[col]
constexpr int BM = 256, BN = 64, BK = 32;

__launch_bounds__(256)
__global__ void gemm_bf16(const u16* __restrict__ A, const u16* __restrict__ Bw,
                          float* __restrict__ outF, u16* __restrict__ outB,
                          const float* __restrict__ bias,
                          const float* __restrict__ residual,
                          int N, int K, int mode) {
  __shared__ u16 As[BM * BK];                 // 16 KB
  __shared__ unsigned BsP[(BK / 2) * BN];     // 4 KB, pair-packed
  const int col0 = blockIdx.x * BN;
  const int row0 = blockIdx.y * BM;
  const int t = threadIdx.x;
  const int w = t >> 5;

  v8f acc[2][4];
#pragma unroll
  for (int i = 0; i < 2; ++i)
#pragma unroll
    for (int j = 0; j < 4; ++j) acc[i][j] = v8f_zero();

  const int ar = t >> 2, ac = (t & 3) * 8;      // A stage: 4 rows per thread
  const int bkp = t >> 4, bn0 = (t & 15) * 4;   // B stage: one packed uint4

  for (int k0 = 0; k0 < K; k0 += BK) {
    // ---- issue ALL staging loads first (independent temps -> MLP) ----
    uint4 at[4];
#pragma unroll
    for (int i = 0; i < 4; ++i)
      at[i] = *(const uint4*)(A + (size_t)(row0 + ar + 64 * i) * K + k0 + ac);
    const u16* s0 = Bw + (size_t)(k0 + 2 * bkp) * N + col0 + bn0;
    uint2 blo = *(const uint2*)s0;              // row 2kp, cols n0..n0+3
    uint2 bhi = *(const uint2*)(s0 + N);        // row 2kp+1
    // ---- then store to LDS ----
#pragma unroll
    for (int i = 0; i < 4; ++i)
      *(uint4*)&As[(ar + 64 * i) * BK + ac] = at[i];
    {
      uint4 o;
      o.x = (blo.x & 0xffffu) | (bhi.x << 16);
      o.y = (blo.x >> 16)     | (bhi.x & 0xffff0000u);
      o.z = (blo.y & 0xffffu) | (bhi.y << 16);
      o.w = (blo.y >> 16)     | (bhi.y & 0xffff0000u);
      *(uint4*)&BsP[bkp * BN + bn0] = o;
    }
    if (k0 + BK < K) {  // global_prefetch_b8 for next K tiles
      __builtin_prefetch(A + (size_t)(row0 + ar) * K + k0 + BK + ac, 0, 1);
      __builtin_prefetch(Bw + (size_t)(k0 + BK + 2 * bkp) * N + col0 + bn0, 0, 1);
    }
    __syncthreads();
    v16bf a0 = load_a_frag(&As[(w * 32) * BK], BK);
    v16bf a1 = load_a_frag(&As[(w * 32 + 16) * BK], BK);
#pragma unroll
    for (int tn = 0; tn < 4; ++tn) {
      v16bf bf = load_bp_frag(&BsP[tn * 16], BN);
      acc[0][tn] = wmma_bf16(a0, bf, acc[0][tn]);
      acc[1][tn] = wmma_bf16(a1, bf, acc[1][tn]);
    }
    __syncthreads();
  }

  const unsigned lane = t & 31u, nlane = lane & 15u, hf = lane >> 4;
#pragma unroll
  for (int sub = 0; sub < 2; ++sub) {
#pragma unroll
    for (int tn = 0; tn < 4; ++tn) {
      int col = col0 + tn * 16 + (int)nlane;
#pragma unroll
      for (int r = 0; r < 8; ++r) {
        int row = row0 + w * 32 + sub * 16 + r + (int)hf * 8;
        size_t idx = (size_t)row * N + col;
        float v = acc[sub][tn][r];
        if (mode == 0) {
          if (col < E_) v *= 0.125f;            // q * 1/sqrt(DH)
          outF[idx] = v;
        } else if (mode == 1) {
          outF[idx] = residual[idx] + v;
        } else if (mode == 2) {
          outB[idx] = f2bf(gelu_f(v + bias[col]));
        } else {
          outF[idx] = residual[idx] + v + bias[col];
        }
      }
    }
  }
}

// ----------------------- block-sparse flash attention ------------------------
// One block per (b, h, query-block). 4 waves; wave w owns rows 16w..16w+15.
// Online softmax over the key blocks selected by the BigBird bitmask.
// All staging is float4 global loads packed into uint2/uint4 LDS stores.
__launch_bounds__(128)
__global__ void bigbird_attn_kernel(const float* __restrict__ qkv,
                                    const int* __restrict__ tokens,
                                    u16* __restrict__ obuf,
                                    const unsigned* __restrict__ maskrows) {
  __shared__ u16 Qs[64 * 64];          // Q block, row-major [q][d]
  __shared__ u16 Ks[64 * 64];          // K block, row-major [key][d] (= K^T [n][k])
  __shared__ unsigned VsP[32 * 64];    // V block, pair-packed [keypair][d]
  __shared__ u16 Ps[64 * 64];          // probs, row-major [q][key]

  const int bid = blockIdx.x;
  const int ib = bid & (NB_ - 1);
  const int h  = (bid >> 5) & (H_ - 1);
  const int b  = bid >> 8;
  const int t  = threadIdx.x;
  const int w  = t >> 5;
  const unsigned lane = t & 31u, nlane = lane & 15u, hf = lane >> 4;

  // stage Q block (already scaled by 1/sqrt(DH) in the QKV GEMM epilogue)
  for (int e = t; e < 1024; e += 128) {           // 64 rows x 16 float4
    int r = e >> 4, d0 = (e & 15) * 4;
    size_t gl = (size_t)(b * L_ + ib * 64 + r);
    float4 q4 = *(const float4*)(qkv + gl * (3 * E_) + h * DH_ + d0);
    uint2 o = { pack2bf(q4.x, q4.y), pack2bf(q4.z, q4.w) };
    *(uint2*)&Qs[r * 64 + d0] = o;
  }

  v8f acc[4];
  float rowm[8], rowl[8];
#pragma unroll
  for (int i = 0; i < 4; ++i) acc[i] = v8f_zero();
#pragma unroll
  for (int r = 0; r < 8; ++r) { rowm[r] = -3.0e38f; rowl[r] = 0.f; }

  const unsigned maskbits = maskrows[ib];

  for (int j = 0; j < NB_; ++j) {
    if (!((maskbits >> j) & 1u)) continue;   // uniform across block
    __syncthreads();
    // stage K block (row-major bf16) — float4 loads, uint2 stores
    for (int e = t; e < 1024; e += 128) {
      int r = e >> 4, d0 = (e & 15) * 4;
      size_t gl = (size_t)(b * L_ + j * 64 + r);
      float4 k4 = *(const float4*)(qkv + gl * (3 * E_) + E_ + h * DH_ + d0);
      uint2 o = { pack2bf(k4.x, k4.y), pack2bf(k4.z, k4.w) };
      *(uint2*)&Ks[r * 64 + d0] = o;
    }
    // stage V block pair-packed: VsP[kp][d] = (V[2kp][d], V[2kp+1][d])
    for (int e = t; e < 512; e += 128) {
      int kp = e >> 4, d0 = (e & 15) * 4;
      size_t gl = (size_t)(b * L_ + j * 64 + 2 * kp);
      const float* vp = qkv + gl * (3 * E_) + 2 * E_ + h * DH_ + d0;
      float4 v0 = *(const float4*)vp;
      float4 v1 = *(const float4*)(vp + 3 * E_);  // next key row
      uint4 o;
      o.x = pack2bf(v0.x, v1.x);
      o.y = pack2bf(v0.y, v1.y);
      o.z = pack2bf(v0.z, v1.z);
      o.w = pack2bf(v0.w, v1.w);
      *(uint4*)&VsP[kp * 64 + d0] = o;
    }
    __syncthreads();

    // S strip (16 x 64): A = Q rows, B = K^T (k contiguous in Ks rows)
    v8f s[4];
#pragma unroll
    for (int i = 0; i < 4; ++i) s[i] = v8f_zero();
    v16bf a0 = load_a_frag(&Qs[w * 16 * 64], 64);       // d = 0..31
    v16bf a1 = load_a_frag(&Qs[w * 16 * 64 + 32], 64);  // d = 32..63
#pragma unroll
    for (int tn = 0; tn < 4; ++tn) {
      v16bf b0 = load_bt_frag(&Ks[(tn * 16) * 64], 64);
      v16bf b1 = load_bt_frag(&Ks[(tn * 16) * 64 + 32], 64);
      s[tn] = wmma_bf16(a0, b0, s[tn]);
      s[tn] = wmma_bf16(a1, b1, s[tn]);
    }

    // key padding mask (tokens <= 0 -> NEG)
#pragma unroll
    for (int tn = 0; tn < 4; ++tn) {
      int kl = b * L_ + j * 64 + tn * 16 + (int)nlane;
      if (tokens[kl] <= 0) {
#pragma unroll
        for (int r = 0; r < 8; ++r) s[tn][r] = -1e9f;
      }
    }

    // online softmax: row max over 4 col tiles + 16-lane butterfly
    float esc[8];
#pragma unroll
    for (int r = 0; r < 8; ++r) {
      float v = fmaxf(fmaxf(s[0][r], s[1][r]), fmaxf(s[2][r], s[3][r]));
#pragma unroll
      for (int off = 8; off; off >>= 1) v = fmaxf(v, __shfl_xor(v, off, 16));
      float mnew = fmaxf(rowm[r], v);
      esc[r] = __expf(rowm[r] - mnew);
      rowm[r] = mnew;
    }
    float rsum[8];
#pragma unroll
    for (int r = 0; r < 8; ++r) rsum[r] = 0.f;
#pragma unroll
    for (int tn = 0; tn < 4; ++tn) {
#pragma unroll
      for (int r = 0; r < 8; ++r) {
        float pv = __expf(s[tn][r] - rowm[r]);
        rsum[r] += pv;
        Ps[(w * 16 + r + (int)hf * 8) * 64 + tn * 16 + (int)nlane] = f2bf(pv);
      }
    }
#pragma unroll
    for (int r = 0; r < 8; ++r) {
      float v = rsum[r];
#pragma unroll
      for (int off = 8; off; off >>= 1) v += __shfl_xor(v, off, 16);
      rowl[r] = rowl[r] * esc[r] + v;
    }
#pragma unroll
    for (int tn = 0; tn < 4; ++tn)
#pragma unroll
      for (int r = 0; r < 8; ++r) acc[tn][r] *= esc[r];

    // acc += P x V  (P staged via per-wave LDS strip; same-wave dscnt dep)
    v16bf pa0 = load_a_frag(&Ps[w * 16 * 64], 64);       // keys 0..31
    v16bf pa1 = load_a_frag(&Ps[w * 16 * 64 + 32], 64);  // keys 32..63
#pragma unroll
    for (int tn = 0; tn < 4; ++tn) {
      v16bf b0 = load_bp_frag(&VsP[tn * 16], 64);           // key pairs 0..15
      v16bf b1 = load_bp_frag(&VsP[16 * 64 + tn * 16], 64); // pairs 16..31
      acc[tn] = wmma_bf16(pa0, b0, acc[tn]);
      acc[tn] = wmma_bf16(pa1, b1, acc[tn]);
    }
  }

  // normalize and emit O (bf16, feeds the wo projection GEMM)
#pragma unroll
  for (int tn = 0; tn < 4; ++tn) {
#pragma unroll
    for (int r = 0; r < 8; ++r) {
      int rl = w * 16 + r + (int)hf * 8;
      size_t gl = (size_t)(b * L_ + ib * 64 + rl);
      float denom = fmaxf(rowl[r], 1e-20f);
      obuf[gl * E_ + h * DH_ + tn * 16 + (int)nlane] = f2bf(acc[tn][r] / denom);
    }
  }
}

// ----------------------------- orchestration --------------------------------

extern "C" void kernel_launch(void* const* d_in, const int* in_sizes, int n_in,
                              void* d_out, int out_size, void* d_ws, size_t ws_size,
                              hipStream_t stream) {
  (void)in_sizes; (void)n_in; (void)out_size; (void)ws_size;

  const int*   tokens = (const int*)  d_in[0];
  const float* embedw = (const float*)d_in[1];
  const float* ln1_s  = (const float*)d_in[2];
  const float* ln1_b  = (const float*)d_in[3];
  const float* wq     = (const float*)d_in[4];
  const float* wk     = (const float*)d_in[5];
  const float* wv     = (const float*)d_in[6];
  const float* wo     = (const float*)d_in[7];
  const float* ln2_s  = (const float*)d_in[8];
  const float* ln2_b  = (const float*)d_in[9];
  const float* w1     = (const float*)d_in[10];
  const float* b1     = (const float*)d_in[11];
  const float* w2     = (const float*)d_in[12];
  const float* b2     = (const float*)d_in[13];
  const float* lnf_s  = (const float*)d_in[14];
  const float* lnf_b  = (const float*)d_in[15];

  char* p = (char*)d_ws;
  auto carve = [&](size_t bytes) -> char* {
    char* r = p;
    p += (bytes + 255) & ~(size_t)255;
    return r;
  };
  const int ROWS = B_ * L_;                 // 8192
  float* x      = (float*)carve((size_t)ROWS * E_ * 4);        // activations (f32)
  u16*   ybf    = (u16*)  carve((size_t)ROWS * E_ * 2);        // LN out (bf16)
  float* qkv    = (float*)carve((size_t)ROWS * 3 * E_ * 4);    // fused qkv (f32)
  u16*   obuf   = (u16*)  carve((size_t)ROWS * E_ * 2);        // attn out (bf16)
  u16*   hbf    = (u16*)  carve((size_t)ROWS * M_ * 2);        // mlp hidden (bf16)
  u16*   wqkvb  = (u16*)  carve((size_t)NL_ * E_ * 3 * E_ * 2);
  u16*   wob    = (u16*)  carve((size_t)NL_ * E_ * E_ * 2);
  u16*   w1b    = (u16*)  carve((size_t)NL_ * E_ * M_ * 2);
  u16*   w2b    = (u16*)  carve((size_t)NL_ * M_ * E_ * 2);
  unsigned* masks = (unsigned*)carve((size_t)NL_ * NB_ * 4);

  build_masks_kernel<<<1, 128, 0, stream>>>(masks);

  // weight conversion to bf16 (wq|wk|wv packed into [E, 3E])
  for (int l = 0; l < NL_; ++l) {
    const int ne = E_ * E_;
    f32_to_bf16_mat<<<(ne + 255) / 256, 256, 0, stream>>>(
        wq + (size_t)l * ne, wqkvb + (size_t)l * E_ * (3 * E_), E_, E_, 3 * E_, 0);
    f32_to_bf16_mat<<<(ne + 255) / 256, 256, 0, stream>>>(
        wk + (size_t)l * ne, wqkvb + (size_t)l * E_ * (3 * E_), E_, E_, 3 * E_, E_);
    f32_to_bf16_mat<<<(ne + 255) / 256, 256, 0, stream>>>(
        wv + (size_t)l * ne, wqkvb + (size_t)l * E_ * (3 * E_), E_, E_, 3 * E_, 2 * E_);
    f32_to_bf16_mat<<<(ne + 255) / 256, 256, 0, stream>>>(
        wo + (size_t)l * ne, wob + (size_t)l * ne, E_, E_, E_, 0);
    const int nm = E_ * M_;
    f32_to_bf16_mat<<<(nm + 255) / 256, 256, 0, stream>>>(
        w1 + (size_t)l * nm, w1b + (size_t)l * nm, E_, M_, M_, 0);
    f32_to_bf16_mat<<<(nm + 255) / 256, 256, 0, stream>>>(
        w2 + (size_t)l * nm, w2b + (size_t)l * nm, M_, E_, E_, 0);
  }

  embed_pe_kernel<<<ROWS, 128, 0, stream>>>(tokens, embedw, x);

  for (int l = 0; l < NL_; ++l) {
    layernorm_kernel<<<ROWS, 128, 0, stream>>>(x, ln1_s + l * E_, ln1_b + l * E_,
                                               nullptr, ybf);
    gemm_bf16<<<dim3((3 * E_) / BN, ROWS / BM), 256, 0, stream>>>(
        ybf, wqkvb + (size_t)l * E_ * (3 * E_), qkv, nullptr, nullptr, nullptr,
        3 * E_, E_, 0);
    bigbird_attn_kernel<<<B_ * H_ * NB_, 128, 0, stream>>>(qkv, tokens, obuf,
                                                           masks + l * NB_);
    gemm_bf16<<<dim3(E_ / BN, ROWS / BM), 256, 0, stream>>>(
        obuf, wob + (size_t)l * E_ * E_, x, nullptr, nullptr, x, E_, E_, 1);
    layernorm_kernel<<<ROWS, 128, 0, stream>>>(x, ln2_s + l * E_, ln2_b + l * E_,
                                               nullptr, ybf);
    gemm_bf16<<<dim3(M_ / BN, ROWS / BM), 256, 0, stream>>>(
        ybf, w1b + (size_t)l * E_ * M_, nullptr, hbf, b1 + l * M_, nullptr,
        M_, E_, 2);
    gemm_bf16<<<dim3(E_ / BN, ROWS / BM), 256, 0, stream>>>(
        hbf, w2b + (size_t)l * M_ * E_, x, nullptr, b2 + l * E_, x, E_, M_, 3);
  }

  layernorm_kernel<<<ROWS, 128, 0, stream>>>(x, lnf_s, lnf_b, (float*)d_out, nullptr);
}
